// WindowAttention_40475771797517
// MI455X (gfx1250) — matrix operand order repaired and checked by reading, not compiled, not run on hardware
//
#include <hip/hip_runtime.h>
#include <hip/hip_bf16.h>

// ============================================================================
// Fused Swin window attention for gfx1250 (MI455X), wave32 + WMMA f16.
//
//   out[b] = softmax( (x W_q)(x W_k)^T * d^-1/2 + bias ) (x W_v) W_proj + b_proj
//
// One workgroup per window (8192 blocks, 256 threads = 8 wave32).
// All GEMMs run on v_wmma_f32_16x16x32_f16. Weights are pre-packed (prep
// kernels) into the CDNA5 WMMA B-operand lane layout so every B operand is a
// single contiguous 32B v16h load from L2-resident memory; x / q / probs / o
// are staged in LDS in the A-operand packed layout. The per-head output is
// projected immediately (out = sum_h o_h @ Wp[h*32:(h+1)*32,:]) and
// accumulated in registers, so nothing but x and out touches HBM.
// ============================================================================

typedef __attribute__((ext_vector_type(16))) _Float16 v16h;
typedef __attribute__((ext_vector_type(8)))  float    v8f;

#define DIMC   256
#define HEADS  8
#define N_TOK  49
#define NP     64          // tokens padded to 4 row-tiles
#define B_WIN  8192
#define TPB    256
#define NEG_BIG (-1e30f)

// ---------------------------------------------------------------------------
// WMMA 16x16x32 f16 operand layouts used here (k order is this code's, not ISA table):
//  A (16x32): lane L -> M = L%16, hi = L/16; vgpr-pair p holds
//             K = 2p + 16*hi  (contiguous order chosen here, same as B)
//  B (32x16): lane L -> N = L%16, hi = L/16; vgpr-pair p holds K = 2p + 16*hi (chosen order)
//  C (16x16 f32): vgpr v -> M = v + 8*(L/16), N = L%16
// a_slot(kk) returns (hi<<4)|slot for K-in-tile kk of the A layout.
// ---------------------------------------------------------------------------
__device__ __host__ __forceinline__ int a_slot(int kk) {
  return kk;
}

// ===================== prep: pack W_qkv into B-layout tiles =================
// dst tile (kt in 0..7 of K=32, ctile in 0..47 of N=16): 512 contiguous f16,
// element [lane*16 + slot] with K = 2*(slot/2) + 16*(lane/16) + (slot&1),
// N = ctile*16 + lane%16.
__global__ void pack_wqkv_kernel(const float* __restrict__ w,
                                 _Float16* __restrict__ dst) {
  int idx = blockIdx.x * blockDim.x + threadIdx.x;    // 8*48*512 total
  if (idx >= 8 * 48 * 512) return;
  int e     = idx & 511;
  int ctile = (idx >> 9) % 48;
  int kt    = idx / (48 * 512);
  int lane = e >> 4, slot = e & 15;
  int k   = kt * 32 + 2 * (slot >> 1) + 16 * (lane >> 4) + (slot & 1);
  int col = ctile * 16 + (lane & 15);
  dst[idx] = (_Float16)w[k * (3 * DIMC) + col];
}

__global__ void pack_wproj_kernel(const float* __restrict__ w,
                                  _Float16* __restrict__ dst) {
  int idx = blockIdx.x * blockDim.x + threadIdx.x;    // 8*16*512 total
  if (idx >= 8 * 16 * 512) return;
  int e     = idx & 511;
  int ctile = (idx >> 9) % 16;
  int kt    = idx / (16 * 512);
  int lane = e >> 4, slot = e & 15;
  int k   = kt * 32 + 2 * (slot >> 1) + 16 * (lane >> 4) + (slot & 1);
  int col = ctile * 16 + (lane & 15);
  dst[idx] = (_Float16)w[k * DIMC + col];
}

// ============== prep: padded bias table biasP[h][64][64] (f32) ==============
// Padded columns get -1e30 so softmax weights vanish there.
__global__ void build_bias_kernel(const float* __restrict__ rel_bias,
                                  const int*   __restrict__ rel_index,
                                  float* __restrict__ biasP) {
  int idx = blockIdx.x * blockDim.x + threadIdx.x;    // 8*64*64 total
  if (idx >= HEADS * NP * NP) return;
  int n = idx & 63, m = (idx >> 6) & 63, h = idx >> 12;
  float v;
  if (n >= N_TOK)      v = NEG_BIG;                   // mask padded keys
  else if (m >= N_TOK) v = 0.f;                       // padded query rows: don't care
  else                 v = rel_bias[rel_index[m * N_TOK + n] * HEADS + h];
  biasP[idx] = v;
}

// =========================== fused attention kernel =========================
__global__ __launch_bounds__(TPB)
void win_attn_kernel(const float*    __restrict__ x,
                     const _Float16* __restrict__ wq_p,
                     const _Float16* __restrict__ wp_p,
                     const float*    __restrict__ b_proj,
                     const float*    __restrict__ biasP,
                     float*          __restrict__ out) {
  __shared__ alignas(32) _Float16 xh[NP * DIMC];   // 32 KB  x, A-packed
  __shared__ alignas(32) _Float16 qh[4 * 32 * 16]; //  4 KB  q (head), A-packed
  __shared__ alignas(32) _Float16 kh[4 * 32 * 16]; //  4 KB  k^T (head), B-packed
  __shared__ alignas(32) _Float16 vh[4 * 32 * 16]; //  4 KB  v (head), B-packed
  __shared__ alignas(32) _Float16 ph[8 * 32 * 16]; //  8 KB  probs, A-packed
  __shared__ alignas(32) _Float16 oh[4 * 32 * 16]; //  4 KB  o (head), A-packed
  __shared__ float redmax[NP * 2];                 // cross-wave softmax partials
  __shared__ float redsum[NP * 2];

  const int tid  = threadIdx.x;
  const int lane = tid & 31;
  const int wv   = tid >> 5;       // wave id 0..7
  const int n    = lane & 15;      // N coordinate in B/C layouts
  const int hic  = lane >> 4;      // C-layout row-half
  const int b    = blockIdx.x;
  const int mt   = wv & 3;         // row-tile owned by this wave
  const int cg   = wv >> 2;        // column group 0/1

  // ---- stage x[b] -> LDS (fp32 -> f16, A-packed, rows 49..63 zeroed) ----
  const float* xb = x + (size_t)b * (N_TOK * DIMC);
  for (int e = tid; e < NP * DIMC; e += TPB) {
    int m = e >> 8, c = e & 255;
    float val = (m < N_TOK) ? xb[m * DIMC + c] : 0.f;
    int s = a_slot(c & 31);
    xh[(((m >> 4) * 8 + (c >> 5)) * 32 + (m & 15) + (s & 16)) * 16 + (s & 15)] =
        (_Float16)val;
  }
  __syncthreads();

  v8f pacc[8];                     // projection accumulators (64 f32/lane)
  #pragma unroll
  for (int j = 0; j < 8; ++j) pacc[j] = {};

  const float scale = 0.17677669529663687f;   // 32^-0.5 folded into q

  for (int h = 0; h < HEADS; ++h) {
    // ------------------- QKV projection (K=256, 8 wmma steps) -------------
    v8f accq = {}, acck = {}, accv = {};
    #pragma unroll
    for (int kt = 0; kt < 8; ++kt) {
      v16h a = *reinterpret_cast<const v16h*>(&xh[((mt * 8 + kt) * 32 + lane) * 16]);
      const _Float16* wb = wq_p + (size_t)(kt * 48) * 512 + lane * 16;
      v16h bq = *reinterpret_cast<const v16h*>(wb + (size_t)(h * 2 + cg) * 512);
      v16h bk = *reinterpret_cast<const v16h*>(wb + (size_t)(16 + h * 2 + cg) * 512);
      v16h bv = *reinterpret_cast<const v16h*>(wb + (size_t)(32 + h * 2 + cg) * 512);
      accq = __builtin_amdgcn_wmma_f32_16x16x32_f16(false, a, false, bq, (short)0, accq, false, false);
      acck = __builtin_amdgcn_wmma_f32_16x16x32_f16(false, a, false, bk, (short)0, acck, false, false);
      accv = __builtin_amdgcn_wmma_f32_16x16x32_f16(false, a, false, bv, (short)0, accv, false, false);
    }
    // C-layout -> packed LDS (q: A-layout, k^T/v: B-layout)
    #pragma unroll
    for (int v = 0; v < 8; ++v) {
      int ml = v + 8 * hic;                      // token row within tile
      int d  = cg * 16 + n;                      // head-dim coordinate
      int s  = a_slot(d);
      qh[(mt * 32 + ml + (s & 16)) * 16 + (s & 15)] = (_Float16)(accq[v] * scale);
      kh[(mt * 32 + ml + 16 * cg) * 16 + n]         = (_Float16)acck[v];
      vh[(((mt >> 1) * 2 + cg) * 32 + n + 16 * (mt & 1)) * 16 + ml] =
          (_Float16)accv[v];
    }
    __syncthreads();

    // ------------------- scores = q k^T + bias, softmax -------------------
    const int nt0 = cg * 2, nt1 = cg * 2 + 1;
    v16h aq  = *reinterpret_cast<const v16h*>(&qh[(mt * 32 + lane) * 16]);
    v16h bk0 = *reinterpret_cast<const v16h*>(&kh[(nt0 * 32 + lane) * 16]);
    v16h bk1 = *reinterpret_cast<const v16h*>(&kh[(nt1 * 32 + lane) * 16]);
    v8f s0 = {}, s1 = {};
    s0 = __builtin_amdgcn_wmma_f32_16x16x32_f16(false, aq, false, bk0, (short)0, s0, false, false);
    s1 = __builtin_amdgcn_wmma_f32_16x16x32_f16(false, aq, false, bk1, (short)0, s1, false, false);

    float p0[8], p1[8], rm[8];
    const float* bh = biasP + (size_t)h * (NP * NP);
    #pragma unroll
    for (int v = 0; v < 8; ++v) {
      int M = mt * 16 + v + 8 * hic;
      p0[v] = s0[v] + bh[M * NP + nt0 * 16 + n];
      p1[v] = s1[v] + bh[M * NP + nt1 * 16 + n];
      rm[v] = fmaxf(p0[v], p1[v]);
    }
    #pragma unroll
    for (int off = 8; off > 0; off >>= 1)
      #pragma unroll
      for (int v = 0; v < 8; ++v)
        rm[v] = fmaxf(rm[v], __shfl_xor(rm[v], off));   // 16-lane row groups
    if (n == 0)
      #pragma unroll
      for (int v = 0; v < 8; ++v)
        redmax[(mt * 16 + v + 8 * hic) * 2 + cg] = rm[v];
    __syncthreads();

    float rs[8];
    #pragma unroll
    for (int v = 0; v < 8; ++v) {
      int r = mt * 16 + v + 8 * hic;
      float mx = fmaxf(redmax[r * 2], redmax[r * 2 + 1]);
      p0[v] = __expf(p0[v] - mx);
      p1[v] = __expf(p1[v] - mx);
      rs[v] = p0[v] + p1[v];
    }
    #pragma unroll
    for (int off = 8; off > 0; off >>= 1)
      #pragma unroll
      for (int v = 0; v < 8; ++v)
        rs[v] += __shfl_xor(rs[v], off);
    if (n == 0)
      #pragma unroll
      for (int v = 0; v < 8; ++v)
        redsum[(mt * 16 + v + 8 * hic) * 2 + cg] = rs[v];
    __syncthreads();

    #pragma unroll
    for (int v = 0; v < 8; ++v) {
      int M = mt * 16 + v + 8 * hic;
      float inv = 1.f / (redsum[M * 2] + redsum[M * 2 + 1]);
      int N0 = nt0 * 16 + n, N1 = nt1 * 16 + n;
      int sA = a_slot(N0 & 31);
      ph[((mt * 2 + (N0 >> 5)) * 32 + (M & 15) + (sA & 16)) * 16 + (sA & 15)] =
          (_Float16)(p0[v] * inv);
      int sB = a_slot(N1 & 31);
      ph[((mt * 2 + (N1 >> 5)) * 32 + (M & 15) + (sB & 16)) * 16 + (sB & 15)] =
          (_Float16)(p1[v] * inv);
    }
    __syncthreads();

    // ------------------- o_h = probs @ v (K=64, 2 wmma) -------------------
    v8f acco = {};
    #pragma unroll
    for (int kt = 0; kt < 2; ++kt) {
      v16h ap = *reinterpret_cast<const v16h*>(&ph[((mt * 2 + kt) * 32 + lane) * 16]);
      v16h bv = *reinterpret_cast<const v16h*>(&vh[((kt * 2 + cg) * 32 + lane) * 16]);
      acco = __builtin_amdgcn_wmma_f32_16x16x32_f16(false, ap, false, bv, (short)0, acco, false, false);
    }
    #pragma unroll
    for (int v = 0; v < 8; ++v) {
      int ml = v + 8 * hic;
      int s  = a_slot(cg * 16 + n);
      oh[(mt * 32 + ml + (s & 16)) * 16 + (s & 15)] = (_Float16)acco[v];
    }
    __syncthreads();

    // --------- partial projection: pacc += o_h @ Wp[h*32:(h+1)*32,:] ------
    v16h ao = *reinterpret_cast<const v16h*>(&oh[(mt * 32 + lane) * 16]);
    const _Float16* wpb = wp_p + (size_t)(h * 16) * 512 + lane * 16;
    #pragma unroll
    for (int j = 0; j < 8; ++j) {
      int ct = cg * 8 + j;
      v16h bp = *reinterpret_cast<const v16h*>(wpb + (size_t)ct * 512);
      pacc[j] = __builtin_amdgcn_wmma_f32_16x16x32_f16(false, ao, false, bp, (short)0, pacc[j], false, false);
    }
    __syncthreads();
  }

  // ------------------- epilogue: out = pacc + b_proj ----------------------
  float* ob = out + (size_t)b * (N_TOK * DIMC);
  #pragma unroll
  for (int j = 0; j < 8; ++j) {
    int ct = cg * 8 + j;
    float bp = b_proj[ct * 16 + n];
    #pragma unroll
    for (int v = 0; v < 8; ++v) {
      int M = mt * 16 + v + 8 * hic;
      if (M < N_TOK) ob[M * DIMC + ct * 16 + n] = pacc[j][v] + bp;
    }
  }
}

// ============================== launcher ====================================
extern "C" void kernel_launch(void* const* d_in, const int* in_sizes, int n_in,
                              void* d_out, int out_size, void* d_ws, size_t ws_size,
                              hipStream_t stream) {
  (void)in_sizes; (void)n_in; (void)out_size; (void)ws_size;
  const float* x        = (const float*)d_in[0];
  const float* w_qkv    = (const float*)d_in[1];
  const float* w_proj   = (const float*)d_in[2];
  const float* b_proj   = (const float*)d_in[3];
  const float* rel_bias = (const float*)d_in[4];
  const int*   rel_idx  = (const int*)d_in[5];

  // workspace layout (~640 KB): packed f16 weights + padded bias table
  char* ws = (char*)d_ws;
  _Float16* wq_p  = (_Float16*)(ws);                 // 8*48*512 f16 = 384 KB
  _Float16* wp_p  = (_Float16*)(ws + 393216);        // 8*16*512 f16 = 128 KB
  float*    biasP = (float*)   (ws + 524288);        // 8*64*64 f32  = 128 KB

  pack_wqkv_kernel <<<(8 * 48 * 512) / 256, 256, 0, stream>>>(w_qkv, wq_p);
  pack_wproj_kernel<<<(8 * 16 * 512) / 256, 256, 0, stream>>>(w_proj, wp_p);
  build_bias_kernel<<<(HEADS * NP * NP) / 256, 256, 0, stream>>>(rel_bias, rel_idx, biasP);

  win_attn_kernel<<<B_WIN, TPB, 0, stream>>>(x, wq_p, wp_p, b_proj, biasP,
                                             (float*)d_out);
}
